// Fp8DenseGeneralOp_1073741824595
// MI455X (gfx1250) — 8-bit matrix operand order repaired and checked by reading, compiled, not run on hardware
//
#include <hip/hip_runtime.h>
#include <stdint.h>

typedef __attribute__((ext_vector_type(16))) int   v16i;
typedef __attribute__((ext_vector_type(8)))  float v8f;
typedef int b128_t __attribute__((vector_size(16)));
typedef int b64_t  __attribute__((vector_size(8)));

#define K_DIM 4096
#define N_DIM 4096
#define BM 128
#define BN 128
#define BK 128

#if __has_builtin(__builtin_amdgcn_global_load_async_to_lds_b128)
__device__ __forceinline__ void async_ld_b128(const unsigned char* g, unsigned lds_off) {
  __builtin_amdgcn_global_load_async_to_lds_b128(
      (__attribute__((address_space(1))) b128_t*)(uintptr_t)g,
      (__attribute__((address_space(3))) b128_t*)(uintptr_t)lds_off,
      0, 0);
}
#else
__device__ __forceinline__ void async_ld_b128(const unsigned char* g, unsigned lds_off) {
  asm volatile("global_load_async_to_lds_b128 %0, %1, off"
               :: "v"(lds_off), "v"(g) : "memory");
}
#endif

#if __has_builtin(__builtin_amdgcn_global_load_async_to_lds_b64)
__device__ __forceinline__ void async_ld_b64(const unsigned char* g, unsigned lds_off) {
  __builtin_amdgcn_global_load_async_to_lds_b64(
      (__attribute__((address_space(1))) b64_t*)(uintptr_t)g,
      (__attribute__((address_space(3))) b64_t*)(uintptr_t)lds_off,
      0, 0);
}
#else
__device__ __forceinline__ void async_ld_b64(const unsigned char* g, unsigned lds_off) {
  asm volatile("global_load_async_to_lds_b64 %0, %1, off"
               :: "v"(lds_off), "v"(g) : "memory");
}
#endif

#if __has_builtin(__builtin_amdgcn_s_wait_asynccnt)
#define WAIT_ASYNC(n) __builtin_amdgcn_s_wait_asynccnt(n)
#else
#define WAIT_ASYNC(n) asm volatile("s_wait_asynccnt %0" :: "n"(n) : "memory")
#endif

__device__ __forceinline__ float clamp448(float x) {
  return fminf(fmaxf(x, -448.0f), 448.0f);
}

#if __has_builtin(__builtin_amdgcn_cvt_pk_fp8_f32)
__device__ __forceinline__ unsigned int pack4_e4m3(float4 f, float inv) {
  float a0 = clamp448(f.x * inv);
  float a1 = clamp448(f.y * inv);
  float a2 = clamp448(f.z * inv);
  float a3 = clamp448(f.w * inv);
  int w = __builtin_amdgcn_cvt_pk_fp8_f32(a0, a1, 0, false);
  w = __builtin_amdgcn_cvt_pk_fp8_f32(a2, a3, w, true);
  return (unsigned int)w;
}
#else
__device__ __forceinline__ unsigned char f32_to_e4m3(float v) {
  unsigned int u = __float_as_uint(v);
  unsigned char s = (unsigned char)((u >> 24) & 0x80u);
  float a = fabsf(v);
  if (!(a == a)) return (unsigned char)(s | 0x7Fu);
  if (a > 448.0f) a = 448.0f;
  if (a == 0.0f) return s;
  unsigned int ab = __float_as_uint(a);
  int e = (int)(ab >> 23) - 127;
  int qe = (e < -6 ? -6 : e) - 3;
  float sc = __uint_as_float((unsigned int)((127 + 23 + qe) << 23));
  float r = (a + sc) - sc;
  if (r == 0.0f) return s;
  unsigned int rb = __float_as_uint(r);
  int re = (int)(rb >> 23) - 127;
  if (re < -6) {
    unsigned char m = (unsigned char)(r * 512.0f);
    return (unsigned char)(s | m);
  }
  if (re > 8) re = 8;
  unsigned char m = (unsigned char)((rb >> 20) & 0x7u);
  return (unsigned char)(s | (unsigned char)((re + 7) << 3) | m);
}
__device__ __forceinline__ unsigned int pack4_e4m3(float4 f, float inv) {
  unsigned int b0 = f32_to_e4m3(f.x * inv);
  unsigned int b1 = f32_to_e4m3(f.y * inv);
  unsigned int b2 = f32_to_e4m3(f.z * inv);
  unsigned int b3 = f32_to_e4m3(f.w * inv);
  return b0 | (b1 << 8) | (b2 << 16) | (b3 << 24);
}
#endif

__global__ void quant_a_kernel(const float* __restrict__ A,
                               const float* __restrict__ scale,
                               unsigned int* __restrict__ Aq, long long n4) {
  float inv = 1.0f / scale[0];
  long long i = (long long)blockIdx.x * blockDim.x + threadIdx.x;
  long long step = (long long)gridDim.x * blockDim.x;
  for (; i < n4; i += step) {
    float4 f = *(const float4*)(A + i * 4);
    Aq[i] = pack4_e4m3(f, inv);
  }
}

__global__ void quant_bt_kernel(const float* __restrict__ B,
                                const float* __restrict__ scale,
                                unsigned char* __restrict__ BT) {
  __shared__ unsigned int tile[64][17];
  float inv = 1.0f / scale[0];
  int k0 = blockIdx.y * 64;
  int n0 = blockIdx.x * 64;
  int tid = threadIdx.x;
#pragma unroll
  for (int i = 0; i < 4; ++i) {
    int r  = i * 16 + (tid >> 4);
    int c4 = tid & 15;
    float4 f = *(const float4*)(B + (size_t)(k0 + r) * N_DIM + n0 + c4 * 4);
    tile[r][c4] = pack4_e4m3(f, inv);
  }
  __syncthreads();
#pragma unroll
  for (int i = 0; i < 4; ++i) {
    int n  = i * 16 + (tid >> 4);
    int kq = tid & 15;
    unsigned int w = 0;
#pragma unroll
    for (int j = 0; j < 4; ++j) {
      int k = kq * 4 + j;
      unsigned int src = tile[k][n >> 2];
      w |= ((src >> ((n & 3) * 8)) & 0xFFu) << (8 * j);
    }
    *(unsigned int*)(BT + (size_t)(n0 + n) * K_DIM + k0 + kq * 4) = w;
  }
}

__global__ __launch_bounds__(256)
void fp8_wmma_gemm_kernel(const unsigned char* __restrict__ Aq,
                          const unsigned char* __restrict__ BTq,
                          const float* __restrict__ s_in,
                          const float* __restrict__ s_k,
                          float* __restrict__ C, int M) {
  __shared__ __align__(16) unsigned char sA[2][BM * 128];
  __shared__ __align__(16) unsigned char sB[2][BN * 128];

  const int tid = threadIdx.x;
  const int m0 = blockIdx.y * BM;
  const int n0 = blockIdx.x * BN;

  const int lrow   = tid >> 1;
  const int jbase  = (tid & 1) * 4;
  const unsigned char* gA = Aq  + (size_t)(m0 + lrow) * K_DIM + jbase * 16;
  const unsigned char* gB = BTq + (size_t)(n0 + lrow) * K_DIM + jbase * 16;

  const int lane = tid & 31;
  const int w  = tid >> 5;
  const int wm = w >> 2;
  const int wn = w & 3;
  const int h  = lane >> 4;
  const int r  = lane & 15;

  v8f acc[4][2];
#pragma unroll
  for (int im = 0; im < 4; ++im)
#pragma unroll
    for (int in = 0; in < 2; ++in) {
      v8f z = {0.f, 0.f, 0.f, 0.f, 0.f, 0.f, 0.f, 0.f};
      acc[im][in] = z;
    }

  const unsigned baseA0 = (unsigned)(uintptr_t)&sA[0][0];
  const unsigned baseB0 = (unsigned)(uintptr_t)&sB[0][0];

  const int wmt = lrow >> 4;
  const int wr  = lrow & 15;
  const int l0  = wr;
  const int l1  = wr + 16;
  const int k0q = (l0 >> 3) & 3;
  const int k1q = (l1 >> 3) & 3;
  const int bsw = (lrow >> 1) & 7;

  auto issue_tile = [&](int kt, int b) {
    const unsigned char* pA = gA + (size_t)kt * BK;
    const unsigned char* pB = gB + (size_t)kt * BK;
    const unsigned tA = baseA0 + (unsigned)b * (BM * 128) + wmt * 2048;
    const unsigned dB = baseB0 + (unsigned)b * (BN * 128) + lrow * 128;
#pragma unroll
    for (int i = 0; i < 4; ++i) {
      int j = jbase + i;
      unsigned s0 = tA + l0 * 64 + (unsigned)(((j >> 1) ^ k0q) & 3) * 16 + (j & 1) * 8;
      unsigned s1 = tA + l1 * 64 + (unsigned)(((j >> 1) ^ k1q) & 3) * 16 + (j & 1) * 8;
      async_ld_b64(pA + i * 16 - (i & 1) * 8, s0);
      async_ld_b64(pA + i * 16 - (i & 1) * 8 + 16, s1);
    }
#pragma unroll
    for (int i = 0; i < 4; ++i) {
      int j = jbase + i;
      unsigned slot = (unsigned)((j ^ bsw) & 7) * 16;
      async_ld_b128(pB + i * 16, dB + slot);
    }
  };

  const int ntiles = K_DIM / BK;
  issue_tile(0, 0);
  int buf = 0;

  const int akey = ((r >> 3) + 2 * h) & 3;

  for (int kt = 0; kt < ntiles; ++kt) {
    if (kt + 1 < ntiles) {
      issue_tile(kt + 1, buf ^ 1);
      WAIT_ASYNC(12);
    } else {
      WAIT_ASYNC(0);
    }
    __syncthreads();

    v16i bfrag[2];
#pragma unroll
    for (int in = 0; in < 2; ++in) {
      const int n = wn * 32 + in * 16 + r;
      const unsigned char* base = &sB[buf][n * 128];
      const int key = (n >> 1) & 7;
      uint4* d = reinterpret_cast<uint4*>(&bfrag[in]);
#pragma unroll
      for (int g = 0; g < 4; ++g)
        d[g] = *(const uint4*)(base + (((2 * g + h) ^ key) & 7) * 16);
    }

#pragma unroll
    for (int im = 0; im < 4; ++im) {
      const unsigned char* base = &sA[buf][(wm * 4 + im) * 2048 + lane * 64];
      v16i afrag;
      uint4* d = reinterpret_cast<uint4*>(&afrag);
#pragma unroll
      for (int q = 0; q < 4; ++q)
        d[q] = *(const uint4*)(base + ((q ^ akey) & 3) * 16);
#pragma unroll
      for (int in = 0; in < 2; ++in)
        acc[im][in] = __builtin_amdgcn_wmma_f32_16x16x128_fp8_fp8(
            afrag, bfrag[in], (short)0, acc[im][in], false, false);
    }

    __syncthreads();
    buf ^= 1;
  }

  const float sc = s_in[0] * s_k[0];
#pragma unroll
  for (int im = 0; im < 4; ++im)
#pragma unroll
    for (int in = 0; in < 2; ++in) {
      const int rbase = m0 + wm * 64 + im * 16 + 8 * h;
      const int col   = n0 + wn * 32 + in * 16 + r;
#pragma unroll
      for (int v = 0; v < 8; ++v)
        C[(size_t)(rbase + v) * N_DIM + col] = acc[im][in][v] * sc;
    }
}

extern "C" void kernel_launch(void* const* d_in, const int* in_sizes, int n_in,
                              void* d_out, int out_size, void* d_ws, size_t ws_size,
                              hipStream_t stream) {
  const float* A    = (const float*)d_in[0];
  const float* B    = (const float*)d_in[1];
  const float* s_in = (const float*)d_in[2];
  const float* s_k  = (const float*)d_in[4];
  float* out = (float*)d_out;

  const int M = in_sizes[0] / K_DIM;

  unsigned char* Aq  = (unsigned char*)d_ws;
  unsigned char* BTq = Aq + (size_t)M * K_DIM;

  long long n4 = ((long long)M * K_DIM) / 4;
  int blocksA = (int)((n4 + 255) / 256);
  if (blocksA > 65535 * 8) blocksA = 65535 * 8;
  quant_a_kernel<<<blocksA, 256, 0, stream>>>(A, s_in, (unsigned int*)Aq, n4);

  dim3 gB(N_DIM / 64, K_DIM / 64);
  quant_bt_kernel<<<gB, 256, 0, stream>>>(B, s_k, BTq);

  dim3 gG(N_DIM / BN, M / BM);
  fp8_wmma_gemm_kernel<<<gG, 256, 0, stream>>>(Aq, BTq, s_in, s_k, out, M);
}
